// RNN_42734924595875
// MI455X (gfx1250) — compile-verified, matrix operand order checked by reading, not run on hardware
//
#include <hip/hip_runtime.h>

// Problem constants (B,T,I,H,O) = (64, 512, 512, 1024, 256)
#define RNN_B 64
#define RNN_T 512
#define RNN_I 512
#define RNN_H 1024
#define RNN_O 256
#define SCAN_BLOCKS (RNN_H / 16)   // 64 workgroups, one 16-col slice each
#define H_ELEMS ((size_t)RNN_B * RNN_H)

typedef __bf16 bf16_t;
typedef __attribute__((ext_vector_type(16))) __bf16 v16bf;
typedef __attribute__((ext_vector_type(8)))  __bf16 bf16x8;
typedef __attribute__((ext_vector_type(8)))  float  v8f;

// ---------------------------------------------------------------------------
// Unified fragment loader for V_WMMA_F32_16X16X32_BF16 (CDNA5 ISA §7.12.2).
// 16-bit A (16x32 MxK) / B (32x16 KxN, mirrored): element (r,k) with
//   lane = (r & 15) + 16*((k>>3)&1),  vgpr v = ((k&7)>>1) + 4*(k>>4)
// => per lane the 16 bf16 are two CONTIGUOUS runs of 8 (k = kh*8+{0..7} and
// k = 16+kh*8+{0..7}), i.e. exactly two b128 loads from row-major bf16.
// ---------------------------------------------------------------------------
static __device__ __forceinline__ v16bf load_frag_bf16(const bf16_t* __restrict__ base,
                                                       size_t ld, int lane) {
    const int r  = lane & 15;
    const int kh = (lane >> 4) & 1;
    const bf16_t* p = base + (size_t)r * ld + kh * 8;
    bf16x8 g0 = *reinterpret_cast<const bf16x8*>(p);
    bf16x8 g1 = *reinterpret_cast<const bf16x8*>(p + 16);
    v16bf f;
#pragma unroll
    for (int i = 0; i < 8; ++i) { f[i] = g0[i]; f[8 + i] = g1[i]; }
    return f;
}

// ---------------------------------------------------------------------------
// prep 1: split W_i2h (H x (I+H)) into bf16 Wx / Wh; zero bf16 h0; zero sync.
// ---------------------------------------------------------------------------
__global__ __launch_bounds__(256) void rnn_prep_w_kernel(const float* __restrict__ W_i2h,
                                                         bf16_t* __restrict__ Wx,
                                                         bf16_t* __restrict__ Wh,
                                                         bf16_t* __restrict__ h0_bf,
                                                         unsigned* __restrict__ sync_ctr) {
    const int idx = blockIdx.x * 256 + threadIdx.x;
    const int total = RNN_H * (RNN_I + RNN_H);
    if (idx == 0) *sync_ctr = 0u;
    if (idx < total) {
        const int r = idx / (RNN_I + RNN_H);
        const int c = idx - r * (RNN_I + RNN_H);
        const float v = W_i2h[idx];
        if (c < RNN_I) Wx[(size_t)r * RNN_I + c] = (bf16_t)v;
        else           Wh[(size_t)r * RNN_H + (c - RNN_I)] = (bf16_t)v;
    }
    if (idx < RNN_B * RNN_H) h0_bf[idx] = (bf16_t)0.0f;   // ping half of hb
}

// prep 2: convert x (B*T*I f32) -> bf16, same layout. 4 elements per thread.
__global__ __launch_bounds__(256) void rnn_prep_x_kernel(const float* __restrict__ x,
                                                         bf16_t* __restrict__ x_bf) {
    const size_t i4 = (size_t)(blockIdx.x * 256 + threadIdx.x) * 4;
    const float4 v = *reinterpret_cast<const float4*>(x + i4);
    bf16_t* o = x_bf + i4;
    o[0] = (bf16_t)v.x; o[1] = (bf16_t)v.y; o[2] = (bf16_t)v.z; o[3] = (bf16_t)v.w;
}

// ---------------------------------------------------------------------------
// Persistent scan kernel: runs all T=512 steps of
//     h_{t+1} = x_t @ Wx^T + b + h_t @ Wh^T
// Grid: 64 blocks x 128 threads (4 waves). Wave w -> batch rows [16w,16w+16),
// block n -> hidden cols [16n,16n+16). Steps separated by a grid-wide barrier
// (monotonic counter, release/acquire fences). f32 h written only at t=T-1.
// Ping-pong = ONE contiguous buffer + integer offset toggle so the compiler
// keeps addrspace(1) inference (global_load_b128, no FLAT/DScnt path).
// ---------------------------------------------------------------------------
__global__ __launch_bounds__(128) void rnn_scan_kernel(const bf16_t* __restrict__ x_bf, // (B,T,I) bf16
                                                       bf16_t* __restrict__ hb,         // (2,B,H) bf16 ping+pong
                                                       const bf16_t* __restrict__ Wx,   // (H,I) bf16
                                                       const bf16_t* __restrict__ Wh,   // (H,H) bf16
                                                       const float*  __restrict__ bias, // (H)
                                                       float* __restrict__ h_final,     // (B,H) f32 (last step)
                                                       unsigned* __restrict__ sync_ctr) {
    const int lane = threadIdx.x & 31;
    const int wave = threadIdx.x >> 5;        // 0..3
    const int mrow = wave * 16;               // batch-row base
    const int ncol = blockIdx.x * 16;         // hidden-col base

    const size_t ldx = (size_t)RNN_T * RNN_I; // batch stride inside x
    const bf16_t* b_x = Wx + (size_t)ncol * RNN_I;
    const bf16_t* b_h = Wh + (size_t)ncol * RNN_H;
    const bf16_t* a_x_base = x_bf + (size_t)mrow * ldx;
    const bf16_t* a_h_base = hb + (size_t)mrow * RNN_H;

    const int n_g = ncol + (lane & 15);
    const int m0  = mrow + ((lane >> 4) << 3);
    const float bv = bias[n_g];

    for (int t = 0; t < RNN_T; ++t) {
        const size_t in_off  = (size_t)(t & 1) * H_ELEMS;        // ping/pong toggle
        const size_t out_off = (size_t)((t + 1) & 1) * H_ELEMS;
        const bf16_t* a_x = a_x_base + (size_t)t * RNN_I;
        const bf16_t* a_h = a_h_base + in_off;

        v8f acc0 = {};
        v8f acc1 = {};

        // x_t @ Wx^T  (K = 512), two chunks/iter on independent accumulators
#pragma unroll 2
        for (int kb = 0; kb < RNN_I; kb += 64) {
            v16bf a0 = load_frag_bf16(a_x + kb,      ldx,   lane);
            v16bf b0 = load_frag_bf16(b_x + kb,      RNN_I, lane);
            acc0 = __builtin_amdgcn_wmma_f32_16x16x32_bf16(false, a0, false, b0,
                                                           (short)0, acc0, false, false);
            v16bf a1 = load_frag_bf16(a_x + kb + 32, ldx,   lane);
            v16bf b1 = load_frag_bf16(b_x + kb + 32, RNN_I, lane);
            acc1 = __builtin_amdgcn_wmma_f32_16x16x32_bf16(false, a1, false, b1,
                                                           (short)0, acc1, false, false);
        }
        // h_t @ Wh^T  (K = 1024)
#pragma unroll 2
        for (int kb = 0; kb < RNN_H; kb += 64) {
            v16bf a0 = load_frag_bf16(a_h + kb,      RNN_H, lane);
            v16bf b0 = load_frag_bf16(b_h + kb,      RNN_H, lane);
            acc0 = __builtin_amdgcn_wmma_f32_16x16x32_bf16(false, a0, false, b0,
                                                           (short)0, acc0, false, false);
            v16bf a1 = load_frag_bf16(a_h + kb + 32, RNN_H, lane);
            v16bf b1 = load_frag_bf16(b_h + kb + 32, RNN_H, lane);
            acc1 = __builtin_amdgcn_wmma_f32_16x16x32_bf16(false, a1, false, b1,
                                                           (short)0, acc1, false, false);
        }

        // C/D layout: VGPR r, lanes 0-15 -> M=r, lanes 16-31 -> M=r+8; N = lane&15.
#pragma unroll
        for (int r = 0; r < 8; ++r) {
            const float hv = acc0[r] + acc1[r] + bv;
            hb[out_off + (size_t)(m0 + r) * RNN_H + n_g] = (bf16_t)hv;
            if (t == RNN_T - 1) {
                h_final[(size_t)(m0 + r) * RNN_H + n_g] = hv;
            }
        }

        // ---- grid-wide barrier between steps ----
        __threadfence();                 // release: make h stores DEV-visible
        __syncthreads();
        if (threadIdx.x == 0) {
            __hip_atomic_fetch_add(sync_ctr, 1u, __ATOMIC_RELAXED,
                                   __HIP_MEMORY_SCOPE_AGENT);
            const unsigned target = (unsigned)(SCAN_BLOCKS * (t + 1));
            while (__hip_atomic_load(sync_ctr, __ATOMIC_RELAXED,
                                     __HIP_MEMORY_SCOPE_AGENT) < target) {
                __builtin_amdgcn_s_sleep(1);
            }
        }
        __syncthreads();
        __threadfence();                 // acquire: drop stale h from near caches
    }
}

// ---------------------------------------------------------------------------
// Epilogue: output = h_T @ W_h2o^T + b_h2o  (64x256, K=1024) — tiny, exact f32.
// ---------------------------------------------------------------------------
__global__ __launch_bounds__(256) void rnn_out_kernel(const float* __restrict__ h,
                                                      const float* __restrict__ W,
                                                      const float* __restrict__ b,
                                                      float* __restrict__ out) {
    const int idx = blockIdx.x * 256 + threadIdx.x;   // 64*256 threads
    const int bm = idx >> 8;
    const int o  = idx & 255;
    const float4* h4 = reinterpret_cast<const float4*>(h + (size_t)bm * RNN_H);
    const float4* w4 = reinterpret_cast<const float4*>(W + (size_t)o * RNN_H);
    float s = 0.0f;
#pragma unroll 4
    for (int k = 0; k < RNN_H / 4; ++k) {
        const float4 av = h4[k];
        const float4 bv = w4[k];
        s += av.x * bv.x + av.y * bv.y + av.z * bv.z + av.w * bv.w;
    }
    out[idx] = s + b[o];
}

__global__ __launch_bounds__(256) void rnn_copy_hidden(const float* __restrict__ h,
                                                       float* __restrict__ out) {
    const int idx = blockIdx.x * 256 + threadIdx.x;   // B*H = 65536 elements
    if (idx < RNN_B * RNN_H) out[idx] = h[idx];
}

// ---------------------------------------------------------------------------
extern "C" void kernel_launch(void* const* d_in, const int* in_sizes, int n_in,
                              void* d_out, int out_size, void* d_ws, size_t ws_size,
                              hipStream_t stream) {
    const float* x     = (const float*)d_in[0];   // (B,T,I)
    const float* W_i2h = (const float*)d_in[1];   // (H, I+H)
    const float* b_i2h = (const float*)d_in[2];   // (H)
    const float* W_h2o = (const float*)d_in[3];   // (O, H)
    const float* b_h2o = (const float*)d_in[4];   // (O)
    float* out = (float*)d_out;                   // [output (B*O) | hidden (B*H)]

    // Workspace (~36 MB): x_bf | Wx | Wh | h_final(f32) | hb(2x) | sync
    char* ws = (char*)d_ws;
    size_t off = 0;
    bf16_t* x_bf    = (bf16_t*)(ws + off); off += (size_t)RNN_B * RNN_T * RNN_I * sizeof(bf16_t);
    bf16_t* Wx      = (bf16_t*)(ws + off); off += (size_t)RNN_H * RNN_I * sizeof(bf16_t);
    bf16_t* Wh      = (bf16_t*)(ws + off); off += (size_t)RNN_H * RNN_H * sizeof(bf16_t);
    float*  h_final = (float*) (ws + off); off += H_ELEMS * sizeof(float);
    bf16_t* hb      = (bf16_t*)(ws + off); off += 2 * H_ELEMS * sizeof(bf16_t);  // contiguous ping+pong
    unsigned* sync_ctr = (unsigned*)(ws + off);

    // 1) weight split/convert + zero bf16 h0 (ping half) + zero barrier counter
    const int prep_elems = RNN_H * (RNN_I + RNN_H);
    rnn_prep_w_kernel<<<(prep_elems + 255) / 256, 256, 0, stream>>>(W_i2h, Wx, Wh, hb, sync_ctr);
    const size_t x_elems = (size_t)RNN_B * RNN_T * RNN_I;
    rnn_prep_x_kernel<<<(unsigned)(x_elems / (4 * 256)), 256, 0, stream>>>(x, x_bf);

    // 2) persistent scan: all 512 steps in one launch, grid-synced internally
    rnn_scan_kernel<<<SCAN_BLOCKS, 128, 0, stream>>>(x_bf, hb, Wx, Wh, b_i2h,
                                                     h_final, sync_ctr);

    // 3) epilogue
    rnn_out_kernel<<<(RNN_B * RNN_O) / 256, 256, 0, stream>>>(h_final, W_h2o, b_h2o, out);
    rnn_copy_hidden<<<(RNN_B * RNN_H) / 256, 256, 0, stream>>>(h_final, out + RNN_B * RNN_O);
}
